// VectorQuantizer_48232482734187
// MI455X (gfx1250) — matrix operand order repaired and checked by reading, not compiled, not run on hardware
//
#include <hip/hip_runtime.h>

// ---------------------------------------------------------------------------
// VectorQuantizer (cosine-sim codebook, EMA update) for gfx1250 / MI455X.
//
// M=32768 tokens, C=8192 codes, D=512.
// dist GEMM = 275 GFLOP -> v_wmma_f32_16x16x32_bf16, fused with argmax so the
// 1 GB dist matrix is never materialized. Non-GEMM stages ~0.3 GB of HBM
// traffic (~15us at 23.3 TB/s). The dist kernel uses one wave per block,
// 32 rows (2 M-tiles) per wave so each B fragment feeds 2 WMMAs (3 ds_load
// per WMMA instead of 4), plus double-buffered B tiles filled with
// GLOBAL_LOAD_ASYNC_TO_LDS_B128 (ASYNCcnt) so tile ct+1 streams while tile
// ct's 32 WMMAs run. Single-wave blocks need no barriers at all.
// ---------------------------------------------------------------------------

typedef __attribute__((ext_vector_type(16))) __bf16 v16bf;
typedef __attribute__((ext_vector_type(8)))  __bf16 v8bf;
typedef __attribute__((ext_vector_type(8)))  float  v8f;
typedef int v4i __attribute__((vector_size(16)));   // matches builtin param type

#define MTOK   32768
#define CCODES 8192
#define DDIM   512
#define DECAYC 0.2f   // (1 - DECAY)

// ---- async global->LDS copy (CDNA5), guarded for toolchain differences ----
#if defined(__gfx1250__) && __has_builtin(__builtin_amdgcn_global_load_async_to_lds_b128)
#define HAVE_ASYNC_LDS 1
#else
#define HAVE_ASYNC_LDS 0
#endif

#if defined(__gfx1250__) && __has_builtin(__builtin_amdgcn_s_wait_asynccnt)
#define WAIT_ASYNC(n) __builtin_amdgcn_s_wait_asynccnt(n)
#elif defined(__gfx1250__)
#define WAIT_ASYNC(n) asm volatile("s_wait_asynccnt " #n)
#else
#define WAIT_ASYNC(n) ((void)0)
#endif

// One 16-code B tile = 16*512 bf16 = 16KB = 1024 b128 chunks = 32 per lane.
#define TILE_CHUNKS 32

__device__ __forceinline__ void load_btile(const __bf16* __restrict__ src,
                                           __bf16* __restrict__ dst, int lane) {
#if HAVE_ASYNC_LDS
#pragma unroll
  for (int j = 0; j < TILE_CHUNKS; ++j) {
    const __bf16* g = src + (size_t)(lane + j * 32) * 8;
    __bf16* l = dst + (size_t)(lane + j * 32) * 8;
    __builtin_amdgcn_global_load_async_to_lds_b128(
        (__attribute__((address_space(1))) v4i*)g,
        (__attribute__((address_space(3))) v4i*)l, 0, 0);
  }
#else
  const v8bf* s = (const v8bf*)src;
  v8bf* d = (v8bf*)dst;
#pragma unroll
  for (int j = 0; j < TILE_CHUNKS; ++j) d[lane + j * 32] = s[lane + j * 32];
#endif
}

// ---------------------------------------------------------------------------
// Row-wise L2 normalize: fp32 copy (EMA stats / gather) + bf16 copy (WMMA).
// ---------------------------------------------------------------------------
__global__ __launch_bounds__(128)
void vq_l2norm_rows(const float* __restrict__ in,
                    float* __restrict__ outf,
                    __bf16* __restrict__ outb) {
  const size_t row = blockIdx.x;
  const int t = threadIdx.x;
  const float* r = in + row * DDIM;

  float vals[4];
  float s = 0.f;
#pragma unroll
  for (int i = 0; i < 4; ++i) { vals[i] = r[t + i * 128]; s += vals[i] * vals[i]; }
#pragma unroll
  for (int m = 16; m >= 1; m >>= 1) s += __shfl_xor(s, m, 32);

  __shared__ float red[4];
  if ((t & 31) == 0) red[t >> 5] = s;
  __syncthreads();
  s = red[0] + red[1] + red[2] + red[3];
  const float inv = 1.0f / fmaxf(sqrtf(s), 1e-12f);

#pragma unroll
  for (int i = 0; i < 4; ++i) {
    const float v = vals[i] * inv;
    outf[row * DDIM + t + i * 128] = v;
    outb[row * DDIM + t + i * 128] = (__bf16)v;
  }
}

// ---------------------------------------------------------------------------
// Fused dist GEMM + argmax. One wave per block, 32 token rows per block.
// A panel (32x512 bf16 = 32KB) resident in LDS; B tiles double-buffered
// (2 x 16KB) and filled asynchronously. Each wave owns its rows end-to-end:
// per-lane running (max, argmax), then a 16-lane shfl_xor reduction matching
// the 16x16 f32 C-layout (VGPR r -> M = r + 8*kh, N = l16).
// ---------------------------------------------------------------------------
__global__ __launch_bounds__(32)
void vq_dist_argmax(const __bf16* __restrict__ Abf,   // [M][D] normalized tokens
                    const __bf16* __restrict__ Bbf,   // [C][D] normalized codes
                    int* __restrict__ ind,            // [M]
                    float* __restrict__ ind_out) {    // [M] (float view in d_out)
  __shared__ __align__(16) __bf16 As[32 * DDIM];      // 32 KB
  __shared__ __align__(16) __bf16 Bs[2][16 * DDIM];   // 2 x 16 KB

  const int lane = threadIdx.x & 31;
  const int kh   = lane >> 4;     // K-half select (WMMA operand layout)
  const int l16  = lane & 15;
  const size_t rowBase = (size_t)blockIdx.x * 32;

  { // A panel: 32 rows contiguous -> 2048 v8bf chunks / 32 lanes
    const v8bf* src = (const v8bf*)(Abf + rowBase * DDIM);
    v8bf* dst = (v8bf*)As;
#pragma unroll
    for (int i = 0; i < 64; ++i) dst[lane + i * 32] = src[lane + i * 32];
  }

  // Prefetch B tile 0 into buffer 0.
  load_btile(Bbf, (__bf16*)Bs[0], lane);

  float best0[8], best1[8];
  int   bidx0[8], bidx1[8];
#pragma unroll
  for (int r = 0; r < 8; ++r) {
    best0[r] = -3.402823466e38f; bidx0[r] = 0;
    best1[r] = -3.402823466e38f; bidx1[r] = 0;
  }

  // A fragment rows for this wave's two M-tiles (contiguous K, matching B:
  // lanes 0-15 hold K[0:16], lanes 16-31 hold K[16:32]).
  const __bf16* arow0 = As + (size_t)l16 * DDIM;
  const __bf16* arow1 = As + (size_t)(16 + l16) * DDIM;

  for (int ct = 0; ct < CCODES / 16; ++ct) {
    const int cur = ct & 1;

    // Current tile must be fully landed in LDS before we read it.
    WAIT_ASYNC(0);
    // Kick off the next tile's transfer; it overlaps the 32 WMMAs below.
    if (ct + 1 < CCODES / 16)
      load_btile(Bbf + (size_t)(ct + 1) * 16 * DDIM, (__bf16*)Bs[cur ^ 1], lane);

    v8f acc0 = {}, acc1 = {};
    const __bf16* bcol = (const __bf16*)Bs[cur] + (size_t)l16 * DDIM + kh * 16;
#pragma unroll
    for (int kt = 0; kt < DDIM / 32; ++kt) {
      // B 32x16 bf16 layout: lane holds 16 contiguous K of column n=l16.
      const v16bf b = *(const v16bf*)(bcol + kt * 32);

      const v8bf a0lo = *(const v8bf*)(arow0 + kt * 32 + kh * 16);
      const v8bf a0hi = *(const v8bf*)(arow0 + kt * 32 + 8 + kh * 16);
      v16bf a0;
#pragma unroll
      for (int i = 0; i < 8; ++i) { a0[i] = a0lo[i]; a0[8 + i] = a0hi[i]; }
      acc0 = __builtin_amdgcn_wmma_f32_16x16x32_bf16(
          false, a0, false, b, (short)0, acc0, false, false);

      const v8bf a1lo = *(const v8bf*)(arow1 + kt * 32 + kh * 16);
      const v8bf a1hi = *(const v8bf*)(arow1 + kt * 32 + 8 + kh * 16);
      v16bf a1;
#pragma unroll
      for (int i = 0; i < 8; ++i) { a1[i] = a1lo[i]; a1[8 + i] = a1hi[i]; }
      acc1 = __builtin_amdgcn_wmma_f32_16x16x32_bf16(
          false, a1, false, b, (short)0, acc1, false, false);
    }

    const int col = ct * 16 + l16;
#pragma unroll
    for (int r = 0; r < 8; ++r) {
      if (acc0[r] > best0[r]) { best0[r] = acc0[r]; bidx0[r] = col; }
      if (acc1[r] > best1[r]) { best1[r] = acc1[r]; bidx1[r] = col; }
    }
  }

  // Reduce over the 16 lanes sharing each row; lowest index wins ties.
#pragma unroll
  for (int r = 0; r < 8; ++r) {
#pragma unroll
    for (int m = 8; m >= 1; m >>= 1) {
      float ov = __shfl_xor(best0[r], m, 16);
      int   oi = __shfl_xor(bidx0[r], m, 16);
      if (ov > best0[r] || (ov == best0[r] && oi < bidx0[r])) {
        best0[r] = ov; bidx0[r] = oi;
      }
      ov = __shfl_xor(best1[r], m, 16);
      oi = __shfl_xor(bidx1[r], m, 16);
      if (ov > best1[r] || (ov == best1[r] && oi < bidx1[r])) {
        best1[r] = ov; bidx1[r] = oi;
      }
    }
  }
  if (l16 == 0) {
#pragma unroll
    for (int r = 0; r < 8; ++r) {
      const size_t row0 = rowBase + kh * 8 + r;        // M-tile 0
      const size_t row1 = rowBase + 16 + kh * 8 + r;   // M-tile 1
      ind[row0] = bidx0[r];
      ind_out[row0] = (float)bidx0[r];
      ind[row1] = bidx1[r];
      ind_out[row1] = (float)bidx1[r];
    }
  }
}

// ---------------------------------------------------------------------------
// Gather quantize = embed[ind] (un-normalized) and scatter EMA statistics:
// bins[c] += 1, embed_sum[c] += flatten (global_atomic_add_f32).
// ---------------------------------------------------------------------------
__global__ __launch_bounds__(128)
void vq_gather_scatter(const float* __restrict__ embed,   // [C][D] raw
                       const float* __restrict__ flat,    // [M][D] normalized fp32
                       const int* __restrict__ ind,       // [M]
                       float* __restrict__ quant_out,     // [M][D]
                       float* __restrict__ embed_sum,     // [C][D]
                       float* __restrict__ bins) {        // [C]
  const size_t row = blockIdx.x;
  const int t = threadIdx.x;
  const int c = ind[row];
  const float* erow = embed + (size_t)c * DDIM;
  const float* frow = flat + row * DDIM;
  float* qrow = quant_out + row * DDIM;
  float* srow = embed_sum + (size_t)c * DDIM;
#pragma unroll
  for (int i = 0; i < 4; ++i) {
    const int d = t + i * 128;
    qrow[d] = erow[d];
    atomicAdd(&srow[d], frow[d]);
  }
  if (t == 0) atomicAdd(&bins[c], 1.0f);
}

// ---------------------------------------------------------------------------
// Per-code EMA update:
//   new_cluster_size = cs + 0.2*(bins - cs)
//   embed_normalized = l2norm(embed_sum / safe_bins)  (or embed_n if bins==0)
//   new_embed        = embed + 0.2*(embed_normalized - embed)
// ---------------------------------------------------------------------------
__global__ __launch_bounds__(128)
void vq_ema_update(const float* __restrict__ embed,      // [C][D] raw
                   const float* __restrict__ embed_n,    // [C][D] normalized
                   const float* __restrict__ embed_sum,  // [C][D]
                   const float* __restrict__ bins,       // [C]
                   const float* __restrict__ cluster_sz, // [C]
                   float* __restrict__ new_embed,        // [C][D]
                   float* __restrict__ new_cs) {         // [C]
  const size_t c = blockIdx.x;
  const int t = threadIdx.x;
  const float b = bins[c];
  const bool zero = (b == 0.0f);
  const float safe = zero ? 1.0f : b;
  const float* srow = embed_sum + c * DDIM;

  float vals[4];
  float s = 0.f;
#pragma unroll
  for (int i = 0; i < 4; ++i) {
    vals[i] = srow[t + i * 128] / safe;
    s += vals[i] * vals[i];
  }
#pragma unroll
  for (int m = 16; m >= 1; m >>= 1) s += __shfl_xor(s, m, 32);
  __shared__ float red[4];
  if ((t & 31) == 0) red[t >> 5] = s;
  __syncthreads();
  s = red[0] + red[1] + red[2] + red[3];
  const float inv = 1.0f / fmaxf(sqrtf(s), 1e-12f);

#pragma unroll
  for (int i = 0; i < 4; ++i) {
    const int d = t + i * 128;
    const float en = zero ? embed_n[c * DDIM + d] : vals[i] * inv;
    const float e  = embed[c * DDIM + d];
    new_embed[c * DDIM + d] = e + DECAYC * (en - e);
  }
  if (t == 0) new_cs[c] = cluster_sz[c] + DECAYC * (b - cluster_sz[c]);
}

// ---------------------------------------------------------------------------
extern "C" void kernel_launch(void* const* d_in, const int* in_sizes, int n_in,
                              void* d_out, int out_size, void* d_ws, size_t ws_size,
                              hipStream_t stream) {
  (void)in_sizes; (void)n_in; (void)out_size; (void)ws_size;

  const float* x          = (const float*)d_in[0];  // (8,4096,512)
  const float* embed      = (const float*)d_in[1];  // (1,8192,512)
  const float* cluster_sz = (const float*)d_in[2];  // (1,8192)

  // Workspace layout (~143 MB)
  char* p = (char*)d_ws;
  float*  flat_f    = (float*)p;   p += (size_t)MTOK * DDIM * sizeof(float);
  __bf16* flat_b    = (__bf16*)p;  p += (size_t)MTOK * DDIM * sizeof(__bf16);
  float*  embn_f    = (float*)p;   p += (size_t)CCODES * DDIM * sizeof(float);
  __bf16* embn_b    = (__bf16*)p;  p += (size_t)CCODES * DDIM * sizeof(__bf16);
  float*  embed_sum = (float*)p;   p += (size_t)CCODES * DDIM * sizeof(float);
  float*  bins      = (float*)p;   p += (size_t)CCODES * sizeof(float);
  int*    ind       = (int*)p;     p += (size_t)MTOK * sizeof(int);

  // Output layout (concatenated, float view): quantize | ind | new_embed | new_cs
  float* out_q  = (float*)d_out;
  float* out_i  = out_q + (size_t)MTOK * DDIM;
  float* out_e  = out_i + (size_t)MTOK;
  float* out_cs = out_e + (size_t)CCODES * DDIM;

  // Zero scatter targets every call (embed_sum and bins are contiguous).
  (void)hipMemsetAsync(embed_sum, 0,
                       (size_t)CCODES * DDIM * sizeof(float) + CCODES * sizeof(float),
                       stream);

  vq_l2norm_rows<<<MTOK,   128, 0, stream>>>(x, flat_f, flat_b);
  vq_l2norm_rows<<<CCODES, 128, 0, stream>>>(embed, embn_f, embn_b);
  vq_dist_argmax<<<MTOK / 32, 32, 0, stream>>>(flat_b, embn_b, ind, out_i);
  vq_gather_scatter<<<MTOK, 128, 0, stream>>>(embed, flat_f, ind, out_q,
                                              embed_sum, bins);
  vq_ema_update<<<CCODES, 128, 0, stream>>>(embed, embn_f, embed_sum, bins,
                                            cluster_sz, out_e, out_cs);
}
